// SAGE_torch_script_21251498180771
// MI455X (gfx1250) — matrix operand order repaired and checked by reading, not compiled, not run on hardware
//
#include <hip/hip_runtime.h>
#include <hip/hip_fp16.h>
#include <math.h>

typedef __attribute__((ext_vector_type(16))) _Float16 v16h;
typedef __attribute__((ext_vector_type(8)))  _Float16 v8h;
typedef __attribute__((ext_vector_type(8)))  float    v8f;
typedef __attribute__((ext_vector_type(4)))  float    v4f;

union F16x16 { v16h v; v8h h[2]; };

#define HDIM   64
#define TILE_M 128
#define EPSN   1e-12f

// ---------------------------------------------------------------------------
// Utility kernels
// ---------------------------------------------------------------------------
__global__ void zero_f32(float* __restrict__ p, int n) {
    int i = blockIdx.x * blockDim.x + threadIdx.x;
    if (i < n) p[i] = 0.0f;
}

// convert 64x64 f32 weight (k-major) to f16 TRANSPOSED (n-major) for B-fragment
// contiguity: dst[n*64+k] = src[k*64+n]
__global__ void cvt_transpose_f16(const float* __restrict__ src,
                                  _Float16* __restrict__ dst) {
    int i = blockIdx.x * blockDim.x + threadIdx.x;   // 4096
    if (i < 4096) {
        int k = i >> 6, n = i & 63;
        dst[n * 64 + k] = (_Float16)src[i];
    }
}

__global__ void degree_kernel(const int* __restrict__ dst,
                              float* __restrict__ deg, int E) {
    int e = blockIdx.x * blockDim.x + threadIdx.x;
    if (e < E) atomicAdd(&deg[dst[e]], 1.0f);
}

__global__ void rdeg_kernel(const float* __restrict__ deg,
                            float* __restrict__ rdeg, int N) {
    int i = blockIdx.x * blockDim.x + threadIdx.x;
    if (i < N) {
        float d = deg[i]; if (d < 1.0f) d = 1.0f;
        rdeg[i] = 1.0f / d;
    }
}

// edge scatter for 64-wide f16 features with packed f16 atomics:
// agg[dst] += x[src], 32 pk_add_f16 atomics per edge
__global__ void scatter_add64h(const _Float16* __restrict__ x,
                               const int* __restrict__ src,
                               const int* __restrict__ dst,
                               _Float16* __restrict__ agg, int E) {
    int e = blockIdx.x * blockDim.x + threadIdx.x;
    if (e >= E) return;
    const __half2* xs = (const __half2*)(x + (long long)src[e] * HDIM);
    __half2*       ad = (__half2*)(agg + (long long)dst[e] * HDIM);
#pragma unroll 8
    for (int f = 0; f < HDIM / 2; ++f) atomicAdd(&ad[f], xs[f]);
}

// scatter-add of 3-wide f32 features (input layer)
__global__ void scatter_add3(const float* __restrict__ x,
                             const int* __restrict__ src,
                             const int* __restrict__ dst,
                             float* __restrict__ agg, int E) {
    int e = blockIdx.x * blockDim.x + threadIdx.x;
    if (e >= E) return;
    const float* xs = x + (long long)src[e] * 3;
    float*       ad = agg + (long long)dst[e] * 3;
    atomicAdd(&ad[0], xs[0]);
    atomicAdd(&ad[1], xs[1]);
    atomicAdd(&ad[2], xs[2]);
}

// ---------------------------------------------------------------------------
// Layer 0: 3 -> 64 SAGEConv, normalize + relu.  One 64-thread block per node.
// ---------------------------------------------------------------------------
__global__ __launch_bounds__(64)
void sage3_kernel(const float* __restrict__ x, const float* __restrict__ agg3,
                  const float* __restrict__ rdeg,
                  const float* __restrict__ Wl, const float* __restrict__ bl,
                  const float* __restrict__ Wr,
                  _Float16* __restrict__ out, int N) {
    int n = blockIdx.x;
    if (n >= N) return;
    int c = threadIdx.x;
    __shared__ float red[64];
    float s = rdeg[n];
    float a0 = agg3[n * 3 + 0] * s, a1 = agg3[n * 3 + 1] * s, a2 = agg3[n * 3 + 2] * s;
    float x0 = x[n * 3 + 0], x1 = x[n * 3 + 1], x2 = x[n * 3 + 2];
    float t = bl[c] + a0 * Wl[c] + a1 * Wl[64 + c] + a2 * Wl[128 + c]
                    + x0 * Wr[c] + x1 * Wr[64 + c] + x2 * Wr[128 + c];
    red[c] = t * t;
    __syncthreads();
    for (int st = 32; st > 0; st >>= 1) {
        if (c < st) red[c] += red[c + st];
        __syncthreads();
    }
    float nrm = fmaxf(sqrtf(red[0]), EPSN);
    out[(long long)n * HDIM + c] = (_Float16)fmaxf(t / nrm, 0.0f); // relu after layer 0
}

// ---------------------------------------------------------------------------
// Main WMMA SAGE layer:  out = (agg*rdeg) @ Wl + bl + x @ Wr  [+norm][+relu]
// Activations f16, weights f16 transposed, f32 accumulate.
// 256 threads = 8 wave32; block tile = 128 rows x 64 cols.
// A-fragments: 2x b128 straight from global (rows consumed once, L2-resident).
// B-fragments: 2x ds_load_b128 from transposed weight tile in LDS.
// mode: bit0 = relu, bit1 = L2 normalize
// ---------------------------------------------------------------------------
__global__ __launch_bounds__(256)
void sage_gemm64_wmma(const _Float16* __restrict__ agg, const _Float16* __restrict__ xin,
                      const float* __restrict__ rdeg,
                      const _Float16* __restrict__ WlT, const _Float16* __restrict__ WrT,
                      const float* __restrict__ bias,
                      _Float16* __restrict__ out, int N, int mode) {
    // LDS: WlT(8K) WrT(8K) sOut(32K f32) bias(256B)
    __shared__ __align__(16) unsigned char smem[16384 + 32768 + 256];
    _Float16 (*sWl)[HDIM] = (_Float16(*)[HDIM])(smem);           // [n][k]
    _Float16 (*sWr)[HDIM] = (_Float16(*)[HDIM])(smem + 8192);
    float    (*sOut)[HDIM] = (float(*)[HDIM])(smem + 16384);
    float*   sB            = (float*)(smem + 16384 + 32768);

    // stage transposed weights (vectorized b128 copies)
    for (int i = threadIdx.x; i < (HDIM * HDIM) / 8; i += 256) {
        ((v8h*)sWl)[i] = ((const v8h*)WlT)[i];
        ((v8h*)sWr)[i] = ((const v8h*)WrT)[i];
    }
    if (threadIdx.x < HDIM) sB[threadIdx.x] = bias[threadIdx.x];
    __syncthreads();

    const int wave = threadIdx.x >> 5;
    const int lane = threadIdx.x & 31;
    const int hs   = lane >> 4;
    const int lr   = lane & 15;
    const int rowBase = wave * 16;

    // this lane's A row (clamped so EXEC stays all-1s for WMMA)
    long long gr = (long long)blockIdx.x * TILE_M + rowBase + lr;
    long long grc = (gr < (long long)N) ? gr : (long long)(N - 1);
    const _Float16* aRow = agg + grc * HDIM;
    const _Float16* xRow = xin + grc * HDIM;
    const _Float16 scl = (_Float16)rdeg[grc];

    v8f acc[4] = {};

#pragma unroll
    for (int ks = 0; ks < 2; ++ks) {
        const int k0 = ks * 32;
        // A fragments: two contiguous 8-half runs per lane
        F16x16 aA, aX;
        aA.h[0] = *(const v8h*)(aRow + k0 + hs * 8);
        aA.h[1] = *(const v8h*)(aRow + k0 + 16 + hs * 8);
        aX.h[0] = *(const v8h*)(xRow + k0 + hs * 8);
        aX.h[1] = *(const v8h*)(xRow + k0 + 16 + hs * 8);
        aA.v = aA.v * scl;                 // fold scatter-mean divide (pk f16 muls)
#pragma unroll
        for (int nt = 0; nt < 4; ++nt) {
            F16x16 bL, bR;
            const _Float16* pl = &sWl[nt * 16 + lr][k0 + hs * 8];
            const _Float16* pr = &sWr[nt * 16 + lr][k0 + hs * 8];
            bL.h[0] = *(const v8h*)(pl);
            bL.h[1] = *(const v8h*)(pl + 16);
            bR.h[0] = *(const v8h*)(pr);
            bR.h[1] = *(const v8h*)(pr + 16);
            acc[nt] = __builtin_amdgcn_wmma_f32_16x16x32_f16(
                false, aA.v, false, bL.v, (short)0, acc[nt], false, false);
            acc[nt] = __builtin_amdgcn_wmma_f32_16x16x32_f16(
                false, aX.v, false, bR.v, (short)0, acc[nt], false, false);
        }
    }

    // exchange accumulators through LDS so one thread sees a whole row
#pragma unroll
    for (int nt = 0; nt < 4; ++nt)
#pragma unroll
        for (int i = 0; i < 8; ++i)
            sOut[rowBase + hs * 8 + i][nt * 16 + lr] = acc[nt][i];
    __syncthreads();

    // ---- fused epilogue: bias -> L2 normalize -> relu -> f16 store ----
    if (threadIdx.x < TILE_M) {
        long long go = (long long)blockIdx.x * TILE_M + threadIdx.x;
        if (go < N) {
            const v4f* rowv = (const v4f*)sOut[threadIdx.x];
            const v4f* bv   = (const v4f*)sB;
            float ss = 0.0f;
#pragma unroll
            for (int i = 0; i < 16; ++i) {
                v4f t = rowv[i] + bv[i];
                ss += t.x * t.x + t.y * t.y + t.z * t.z + t.w * t.w;
            }
            float inv = 1.0f;
            if (mode & 2) inv = 1.0f / fmaxf(sqrtf(ss), EPSN);
#pragma unroll
            for (int i = 0; i < 8; ++i) {
                v4f t0 = (rowv[2 * i]     + bv[2 * i])     * inv;
                v4f t1 = (rowv[2 * i + 1] + bv[2 * i + 1]) * inv;
                v8h o;
#pragma unroll
                for (int j = 0; j < 4; ++j) {
                    float u0 = t0[j], u1 = t1[j];
                    if (mode & 1) { u0 = fmaxf(u0, 0.0f); u1 = fmaxf(u1, 0.0f); }
                    o[j]     = (_Float16)u0;
                    o[4 + j] = (_Float16)u1;
                }
                *(v8h*)(out + go * HDIM + i * 8) = o;
            }
        }
    }
}

// ---------------------------------------------------------------------------
// Policy last layer: 64 -> 1, no norm/relu (f16 activations in, f32 out)
// ---------------------------------------------------------------------------
__global__ void pol_final(const _Float16* __restrict__ xin, const _Float16* __restrict__ agg,
                          const float* __restrict__ rdeg,
                          const float* __restrict__ Wl, const float* __restrict__ bl,
                          const float* __restrict__ Wr,
                          float* __restrict__ pi, int N) {
    int n = blockIdx.x * blockDim.x + threadIdx.x;
    if (n >= N) return;
    float sdeg = rdeg[n];
    float s = bl[0];
    const _Float16* a = agg + (long long)n * HDIM;
    const _Float16* x = xin + (long long)n * HDIM;
#pragma unroll 4
    for (int k = 0; k < HDIM; ++k)
        s += ((float)a[k] * sdeg) * Wl[k] + (float)x[k] * Wr[k];
    pi[n] = s;
}

// ---------------------------------------------------------------------------
// Graph pooling: [sum | max | min | mean] -> gp[G,256]. One 64-thread block/graph.
// ---------------------------------------------------------------------------
__global__ __launch_bounds__(64)
void pool_kernel(const _Float16* __restrict__ ve, const int* __restrict__ bp,
                 float* __restrict__ gp, int G) {
    int g = blockIdx.x;
    if (g >= G) return;
    int f = threadIdx.x;
    int s = bp[g], e = bp[g + 1];
    float sum = 0.0f, mx = -3.4e38f, mn = 3.4e38f;
    for (int n = s; n < e; ++n) {
        float v = (float)ve[(long long)n * HDIM + f];
        sum += v; mx = fmaxf(mx, v); mn = fminf(mn, v);
    }
    float cnt = (float)(e - s);
    gp[g * 256 + f]       = sum;
    gp[g * 256 + 64 + f]  = mx;
    gp[g * 256 + 128 + f] = mn;
    gp[g * 256 + 192 + f] = sum / cnt;
}

// ---------------------------------------------------------------------------
// Value MLP: 256 -> 32 (relu) -> 1 (tanh). One 32-thread block per graph.
// ---------------------------------------------------------------------------
__global__ __launch_bounds__(32)
void mlp_kernel(const float* __restrict__ gp,
                const float* __restrict__ W1, const float* __restrict__ b1,
                const float* __restrict__ W2, const float* __restrict__ b2,
                float* __restrict__ val, int G) {
    int g = blockIdx.x;
    if (g >= G) return;
    int h = threadIdx.x;
    __shared__ float hid[32];
    float s = b1[h];
    const float* row = gp + (long long)g * 256;
#pragma unroll 4
    for (int i = 0; i < 256; ++i) s += row[i] * W1[i * 32 + h];
    hid[h] = fmaxf(s, 0.0f);
    __syncthreads();
    if (h == 0) {
        float v = b2[0];
        for (int j = 0; j < 32; ++j) v += hid[j] * W2[j];
        val[g] = tanhf(v);
    }
}

// ---------------------------------------------------------------------------
// Assemble outputs: pi_out[N-2G], value[G], graph_indices[N-2G], batch_ptr[G+1]
// ---------------------------------------------------------------------------
__global__ void write_out(const float* __restrict__ pi, const float* __restrict__ val,
                          const int* __restrict__ gi, const int* __restrict__ bp,
                          float* __restrict__ out, int N, int G) {
    int n = blockIdx.x * blockDim.x + threadIdx.x;
    int NP = N - 2 * G;
    if (n < N) {
        int g = gi[n];
        int s = bp[g];
        int l = n - s;
        if (l >= 2) {
            int idx = s - 2 * g + (l - 2);
            out[idx] = pi[n];                       // pi_out
            out[NP + G + idx] = (float)g;           // out_graph_indices
        }
    }
    if (n < G)  out[NP + n] = val[n];               // value
    if (n <= G) out[NP + G + NP + n] = (float)(bp[n] - 2 * n);  // out_batch_ptr
}

// ---------------------------------------------------------------------------
// Host-side orchestration
// ---------------------------------------------------------------------------
static inline void zero_async(float* p, int n, hipStream_t s) {
    zero_f32<<<(n + 255) / 256, 256, 0, s>>>(p, n);
}

extern "C" void kernel_launch(void* const* d_in, const int* in_sizes, int n_in,
                              void* d_out, int out_size, void* d_ws, size_t ws_size,
                              hipStream_t stream) {
    const int N = in_sizes[0] / 3;          // x: [N,3]
    const int E = in_sizes[1] / 2;          // edge_index: [2,E]
    const int G = in_sizes[3] - 1;          // batch_ptr: [G+1]

    const float* x_in = (const float*)d_in[0];
    const int*   src  = (const int*)d_in[1];
    const int*   dst  = ((const int*)d_in[1]) + E;
    const int*   gi   = (const int*)d_in[2];
    const int*   bp   = (const int*)d_in[3];

    // gnn layer i -> inputs {4+3i, 5+3i, 6+3i}; value 28..33; policy 34..39; mlp 40..43

    // ---- workspace layout ----
    size_t NH = (size_t)N * HDIM;
    _Float16* bufA = (_Float16*)d_ws;               // f16 activations [N,64]
    _Float16* bufB = bufA + NH;
    _Float16* bufC = bufB + NH;
    _Float16* aggh = bufC + NH;                     // f16 scatter accumulator [N,64]
    float* agg3 = (float*)(aggh + NH);              // [N,3] f32
    float* deg  = agg3 + (size_t)N * 3;
    float* rdg  = deg + N;
    float* pi   = rdg + N;
    float* gp   = pi + N;                           // [G,256]
    float* val  = gp + (size_t)G * 256;
    _Float16* wfT = (_Float16*)(val + G);           // 20 transposed f16 matrices

    const int eb = (E + 255) / 256;
    const int nb = (N + 255) / 256;
    const int gemmB = (N + TILE_M - 1) / TILE_M;

    // ---- convert + transpose the 20 (64x64) weight matrices to f16 ----
    {
        int srcIdx[20], slot = 0;
        for (int i = 1; i <= 7; ++i) { srcIdx[slot++] = 4 + 3 * i; srcIdx[slot++] = 6 + 3 * i; }
        srcIdx[slot++] = 28; srcIdx[slot++] = 30;    // value layer 0
        srcIdx[slot++] = 31; srcIdx[slot++] = 33;    // value layer 1
        srcIdx[slot++] = 34; srcIdx[slot++] = 36;    // policy layer 0
        for (int m = 0; m < 20; ++m)
            cvt_transpose_f16<<<16, 256, 0, stream>>>((const float*)d_in[srcIdx[m]],
                                                      wfT + (size_t)m * 4096);
    }

    // ---- degrees / reciprocal degrees (fixed across layers) ----
    zero_async(deg, N, stream);
    degree_kernel<<<eb, 256, 0, stream>>>(dst, deg, E);
    rdeg_kernel<<<nb, 256, 0, stream>>>(deg, rdg, N);

    // ---- layer 0 (3 -> 64, f32 math, f16 out) ----
    zero_async(agg3, N * 3, stream);
    scatter_add3<<<eb, 256, 0, stream>>>(x_in, src, dst, agg3, E);
    sage3_kernel<<<N, 64, 0, stream>>>(x_in, agg3, rdg,
                                       (const float*)d_in[4], (const float*)d_in[5],
                                       (const float*)d_in[6], bufA, N);

    // ---- gnn layers 1..7 (WMMA), ping-pong bufA<->bufB ----
    _Float16* cur = bufA; _Float16* nxt = bufB;
    for (int i = 1; i <= 7; ++i) {
        zero_async((float*)aggh, N * (HDIM / 2), stream);
        scatter_add64h<<<eb, 256, 0, stream>>>(cur, src, dst, aggh, E);
        int mode = 2 | (i < 7 ? 1 : 0);   // normalize always; relu except last
        sage_gemm64_wmma<<<gemmB, 256, 0, stream>>>(
            aggh, cur, rdg,
            wfT + (size_t)(2 * (i - 1)) * 4096, wfT + (size_t)(2 * (i - 1) + 1) * 4096,
            (const float*)d_in[4 + 3 * i + 1], nxt, N, mode);
        _Float16* t = cur; cur = nxt; nxt = t;
    }
    _Float16* embeds = cur;               // bufB after 7 swaps

    // ---- policy head ----
    zero_async((float*)aggh, N * (HDIM / 2), stream);
    scatter_add64h<<<eb, 256, 0, stream>>>(embeds, src, dst, aggh, E);
    sage_gemm64_wmma<<<gemmB, 256, 0, stream>>>(
        aggh, embeds, rdg, wfT + 18 * 4096, wfT + 19 * 4096,
        (const float*)d_in[35], bufC, N, /*relu only*/1);
    zero_async((float*)aggh, N * (HDIM / 2), stream);
    scatter_add64h<<<eb, 256, 0, stream>>>(bufC, src, dst, aggh, E);
    pol_final<<<nb, 256, 0, stream>>>(bufC, aggh, rdg,
                                      (const float*)d_in[37], (const float*)d_in[38],
                                      (const float*)d_in[39], pi, N);

    // ---- value head ----
    _Float16* vtmp = (embeds == bufB) ? bufA : bufB;
    zero_async((float*)aggh, N * (HDIM / 2), stream);
    scatter_add64h<<<eb, 256, 0, stream>>>(embeds, src, dst, aggh, E);
    sage_gemm64_wmma<<<gemmB, 256, 0, stream>>>(
        aggh, embeds, rdg, wfT + 14 * 4096, wfT + 15 * 4096,
        (const float*)d_in[29], vtmp, N, /*relu only*/1);
    zero_async((float*)aggh, N * (HDIM / 2), stream);
    scatter_add64h<<<eb, 256, 0, stream>>>(vtmp, src, dst, aggh, E);
    sage_gemm64_wmma<<<gemmB, 256, 0, stream>>>(
        aggh, vtmp, rdg, wfT + 16 * 4096, wfT + 17 * 4096,
        (const float*)d_in[32], bufC, N, /*no relu, no norm*/0);

    // ---- pooling + MLP ----
    pool_kernel<<<G, 64, 0, stream>>>(bufC, bp, gp, G);
    mlp_kernel<<<G, 32, 0, stream>>>(gp, (const float*)d_in[40], (const float*)d_in[41],
                                     (const float*)d_in[42], (const float*)d_in[43], val, G);

    // ---- assemble outputs ----
    write_out<<<nb, 256, 0, stream>>>(pi, val, gi, bp, (float*)d_out, N, G);
    (void)n_in; (void)out_size; (void)ws_size;
}
